// L1Model_30726196035931
// MI455X (gfx1250) — matrix operand order repaired and checked by reading, not compiled, not run on hardware
//
#include <hip/hip_runtime.h>
#include <hip/hip_bf16.h>
#include <math.h>

// ---------------------------------------------------------------------------
// GPT-2 style forward (L=4, D=1024, H=16, F=4096, V=50257, S=2048) for
// MI455X / gfx1250.  All GEMMs use v_wmma_f32_16x16x32_f16 (wave32 WMMA),
// activations f16, f32 accumulate, residual stream f32.  Double-buffered LDS
// staging with async global->LDS copies (ASYNCcnt) where no conversion is
// needed; f32->f16 weight conversion rides the VALU staging path with packed
// b128 LDS stores.
// ---------------------------------------------------------------------------

typedef __attribute__((ext_vector_type(16))) _Float16 v16h;
typedef __attribute__((ext_vector_type(8)))  float    v8f;
typedef int v4i_b __attribute__((vector_size(4 * sizeof(int))));

#define S_LEN 2048
#define D_DIM 1024
#define H_NUM 16
#define HS_DIM 64
#define F_DIM 4096
#define V_DIM 50257
#define L_NUM 4

#if defined(__has_builtin)
#if __has_builtin(__builtin_amdgcn_global_load_async_to_lds_b128)
#define HAVE_ASYNC_COPY 1
#endif
#endif
#ifndef HAVE_ASYNC_COPY
#define HAVE_ASYNC_COPY 0
#endif

// 16-byte global -> LDS copy; async (no VGPR round trip) when available.
__device__ __forceinline__ void async_copy16(const _Float16* g, _Float16* l) {
#if HAVE_ASYNC_COPY
  __builtin_amdgcn_global_load_async_to_lds_b128((v4i_b*)g, (v4i_b*)l, 0, 0);
#else
  *(uint4*)l = *(const uint4*)g;
#endif
}

__device__ __forceinline__ void wait_async() {
#if HAVE_ASYNC_COPY
  asm volatile("s_wait_asynccnt 0" ::: "memory");
#endif
}

// Intra-wave LDS producer->consumer ordering (CDNA5 split counters)
__device__ __forceinline__ void wave_lds_sync() {
  asm volatile("s_wait_dscnt 0" ::: "memory");
}

__device__ __forceinline__ v16h ld_frag16(const _Float16* p0, const _Float16* p1) {
  union { uint4 u[2]; v16h h; } t;
  t.u[0] = *(const uint4*)p0;
  t.u[1] = *(const uint4*)p1;
  return t.h;
}

__device__ __forceinline__ v8f wmma_f16(v16h a, v16h b, v8f c) {
  // D = A(16x32 f16) * B(32x16 f16) + C(16x16 f32)
  return __builtin_amdgcn_wmma_f32_16x16x32_f16(false, a, false, b, (short)0, c,
                                                false, false);
}

// ---------------------------------------------------------------------------
// Embedding: h[s,d] = tok_emb[ids[s], d] + pos_emb[s, d]
// ---------------------------------------------------------------------------
__global__ __launch_bounds__(256) void embed_kernel(const int* __restrict__ ids,
                                                    const float* __restrict__ tok,
                                                    const float* __restrict__ pos,
                                                    float* __restrict__ h) {
  int i = blockIdx.x * 256 + threadIdx.x;   // over S*D
  int s = i >> 10;
  int d = i & (D_DIM - 1);
  h[i] = tok[(size_t)ids[s] * D_DIM + d] + pos[i];
}

// ---------------------------------------------------------------------------
// LayerNorm over D=1024, one block per row, f16 output (WMMA input)
// ---------------------------------------------------------------------------
__global__ __launch_bounds__(256) void ln_kernel(const float* __restrict__ x,
                                                 const float* __restrict__ w,
                                                 const float* __restrict__ b,
                                                 _Float16* __restrict__ out) {
  __shared__ float red[256];
  const int row = blockIdx.x;
  const float* xr = x + (size_t)row * D_DIM;

  float s = 0.0f;
  for (int d = threadIdx.x; d < D_DIM; d += 256) s += xr[d];
  red[threadIdx.x] = s;
  __syncthreads();
  for (int st = 128; st > 0; st >>= 1) {
    if (threadIdx.x < st) red[threadIdx.x] += red[threadIdx.x + st];
    __syncthreads();
  }
  const float mean = red[0] * (1.0f / D_DIM);
  __syncthreads();

  float vs = 0.0f;
  for (int d = threadIdx.x; d < D_DIM; d += 256) {
    float t = xr[d] - mean;
    vs += t * t;
  }
  red[threadIdx.x] = vs;
  __syncthreads();
  for (int st = 128; st > 0; st >>= 1) {
    if (threadIdx.x < st) red[threadIdx.x] += red[threadIdx.x + st];
    __syncthreads();
  }
  const float rstd = rsqrtf(red[0] * (1.0f / D_DIM) + 1e-5f);

  for (int d = threadIdx.x; d < D_DIM; d += 256)
    out[(size_t)row * D_DIM + d] = (_Float16)((xr[d] - mean) * rstd * w[d] + b[d]);
}

// ---------------------------------------------------------------------------
// WMMA GEMM: C[M,N] = act(A_f16[M,K] @ B_f32 + bias)  (+ optional residual)
//   MODE 0: store f16         MODE 1: store f32        MODE 2: f32 += (residual)
//   BT:   B stored [N,K] (lm_head) instead of [K,N]
// Block tile 128x128, K-step 32, ping-pong double-buffered LDS.
// 8 waves, each wave computes 32x64 (8 WMMA tiles).
// ---------------------------------------------------------------------------
template <int MODE, bool GELU_F, bool BT>
__global__ __launch_bounds__(256) void gemm_kernel(const _Float16* __restrict__ A,
                                                   const float* __restrict__ B,
                                                   const float* __restrict__ bias,
                                                   _Float16* __restrict__ out16,
                                                   float* __restrict__ out32,
                                                   int M, int N, int K) {
  __shared__ __align__(16) _Float16 sA[2][128][40];   // [buf][m][k], pad 8
  __shared__ __align__(16) _Float16 sB[2][32][136];   // [buf][k][n], pad 8

  const int tid  = threadIdx.x;
  const int lane = tid & 31;
  const int wid  = tid >> 5;
  const int lo   = lane & 15;
  const int hi   = lane >> 4;
  const int wM   = wid >> 1;            // 0..3  (32 rows each)
  const int wN   = wid & 1;             // 0..1  (64 cols each)
  const int m0   = blockIdx.y * 128;
  const int n0   = blockIdx.x * 128;

  v8f acc[2][4];
#pragma unroll
  for (int i = 0; i < 2; ++i)
#pragma unroll
    for (int j = 0; j < 4; ++j)
#pragma unroll
      for (int e = 0; e < 8; ++e) acc[i][j][e] = 0.0f;

  // staging indices
  const int arow = tid >> 1;            // 0..127
  const int acg  = (tid & 1) * 16;      // 0 | 16
  const int brow = tid >> 3;            // k 0..31     (!BT)
  const int bcol = (tid & 7) * 16;      // n offset    (!BT)
  const int bn   = tid >> 1;            // n 0..127    (BT)
  const int bkg  = (tid & 1) * 16;      // k 0 | 16    (BT)

  auto stageA = [&](int buf, int k0) {
    const _Float16* ga = A + (size_t)(m0 + arow) * K + k0 + acg;
    async_copy16(ga,     &sA[buf][arow][acg]);
    async_copy16(ga + 8, &sA[buf][arow][acg + 8]);
  };
  auto stageB = [&](int buf, int k0) {
    if (!BT) {
      union { _Float16 h[16]; uint4 u[2]; } pk;
      if (n0 + 128 <= N) {
        const float* gb = B + (size_t)(k0 + brow) * N + n0 + bcol;
#pragma unroll
        for (int j = 0; j < 16; j += 4) {
          float4 f = *(const float4*)(gb + j);
          pk.h[j + 0] = (_Float16)f.x;
          pk.h[j + 1] = (_Float16)f.y;
          pk.h[j + 2] = (_Float16)f.z;
          pk.h[j + 3] = (_Float16)f.w;
        }
      } else {
#pragma unroll
        for (int j = 0; j < 16; ++j) {
          int n = n0 + bcol + j;
          pk.h[j] = (_Float16)((n < N) ? B[(size_t)(k0 + brow) * N + n] : 0.0f);
        }
      }
      // packed commit: 2 x ds_store_b128 instead of 16 x ds_store_b16
      *(uint4*)&sB[buf][brow][bcol]     = pk.u[0];
      *(uint4*)&sB[buf][brow][bcol + 8] = pk.u[1];
    } else {  // B stored [N,K]; element(k,n) = B[n*K + k] -> column scatter
      const int n = n0 + bn;
      if (n < N) {
        const float* gb = B + (size_t)n * K + k0 + bkg;
#pragma unroll
        for (int j = 0; j < 16; ++j) sB[buf][bkg + j][bn] = (_Float16)gb[j];
      } else {
#pragma unroll
        for (int j = 0; j < 16; ++j) sB[buf][bkg + j][bn] = (_Float16)0.0f;
      }
    }
  };

  const int nIter = K >> 5;
  stageA(0, 0);
  stageB(0, 0);

  for (int it = 0; it < nIter; ++it) {
    const int cur = it & 1;
    const int k0  = it << 5;

    wait_async();                                   // our async A copies done
    asm volatile("s_wait_dscnt 0" ::: "memory");    // our B ds_stores done
    __syncthreads();                                // tile visible to all waves

    if (it + 1 < nIter) {
      stageA(cur ^ 1, k0 + 32);                     // overlap next tile with math
      stageB(cur ^ 1, k0 + 32);
      if (!BT && it + 2 < nIter)                    // pre-warm GL2 for weights
        __builtin_prefetch(B + (size_t)(k0 + 64 + brow) * N + n0 + bcol, 0, 1);
      if (BT && it + 2 < nIter && n0 + bn < N)
        __builtin_prefetch(B + (size_t)(n0 + bn) * K + k0 + 64 + bkg, 0, 1);
    }

    // ---- fragments + 8 WMMAs from buffer `cur` ----
    v16h af[2], bf[4];
#pragma unroll
    for (int mi = 0; mi < 2; ++mi) {
      int r = wM * 32 + mi * 16 + lo;
      af[mi] = ld_frag16(&sA[cur][r][hi * 8], &sA[cur][r][16 + hi * 8]);
    }
#pragma unroll
    for (int ni = 0; ni < 4; ++ni) {
      int c  = wN * 64 + ni * 16 + lo;
#pragma unroll
      for (int s = 0; s < 16; ++s) {
        int kr = (s < 8) ? hi * 8 + s : 16 + hi * 8 + (s - 8);
        bf[ni][s] = sB[cur][kr][c];
      }
    }
#pragma unroll
    for (int mi = 0; mi < 2; ++mi)
#pragma unroll
      for (int ni = 0; ni < 4; ++ni)
        acc[mi][ni] = wmma_f16(af[mi], bf[ni], acc[mi][ni]);
  }

  // ---- epilogue: bias (+GELU) (+residual) ----
#pragma unroll
  for (int mi = 0; mi < 2; ++mi) {
#pragma unroll
    for (int ni = 0; ni < 4; ++ni) {
      const int n = n0 + wN * 64 + ni * 16 + lo;
      if (n >= N) continue;
      const float bv = bias ? bias[n] : 0.0f;
#pragma unroll
      for (int r = 0; r < 8; ++r) {
        const int m = m0 + wM * 32 + mi * 16 + hi * 8 + r;
        if (m >= M) continue;
        float v = acc[mi][ni][r] + bv;
        if (GELU_F) v = 0.5f * v * (1.0f + erff(v * 0.70710678f));
        const size_t idx = (size_t)m * N + n;
        if (MODE == 0)      out16[idx] = (_Float16)v;
        else if (MODE == 1) out32[idx] = v;
        else                out32[idx] += v;  // residual accumulate
      }
    }
  }
}

// ---------------------------------------------------------------------------
// Flash-style causal attention with WMMA. Grid: (S/128, H).
// 8 waves/block; wave w owns 16 query rows. Streams 32-key blocks through LDS.
// ---------------------------------------------------------------------------
__global__ __launch_bounds__(256) void attn_kernel(const _Float16* __restrict__ q,
                                                   const _Float16* __restrict__ k,
                                                   const _Float16* __restrict__ v,
                                                   _Float16* __restrict__ o) {
  __shared__ __align__(16) _Float16 sKt[64][32];      // [hs][key] (transposed)
  __shared__ __align__(16) _Float16 sVt[32][64];      // [key][hs]
  __shared__ float    sS[8][16][33];                  // per-wave score scratch
  __shared__ __align__(16) _Float16 sP[8][16][32];    // per-wave probs (A-frag)
  __shared__ float    sCorr[8][16];
  __shared__ float    sL[8][16];

  const int tid  = threadIdx.x;
  const int lane = tid & 31;
  const int wid  = tid >> 5;
  const int lo   = lane & 15;
  const int hi   = lane >> 4;
  const int head = blockIdx.y;
  const int qblk = blockIdx.x;
  const int q0   = qblk * 128 + wid * 16;             // wave's query base
  const size_t hoff = (size_t)head * HS_DIM;

  // Q fragments (two 16x32 K-slices of the 16x64 Q tile)
  v16h aQ[2];
#pragma unroll
  for (int ks = 0; ks < 2; ++ks) {
    const _Float16* gq = q + (size_t)(q0 + lo) * D_DIM + hoff + ks * 32 + hi * 8;
    union { uint4 u[2]; v16h h; } t;
    t.u[0] = *(const uint4*)gq;
    t.u[1] = *(const uint4*)(gq + 16);
    aQ[ks] = t.h;
  }

  v8f O[4];
#pragma unroll
  for (int ni = 0; ni < 4; ++ni)
#pragma unroll
    for (int e = 0; e < 8; ++e) O[ni][e] = 0.0f;

  float rm = -1e30f, rl = 0.0f;   // lanes 0..15: running max/sum of row `lane`

  const int kbEnd   = qblk * 4 + 4;           // block-wide key-block count
  const int myKbEnd = (q0 + 15) / 32 + 1;     // wave's causal limit

  const int lkey = tid >> 3;                  // 0..31
  const int lhs  = (tid & 7) * 8;             // 0..56

  for (int kb = 0; kb < kbEnd; ++kb) {
    __syncthreads();
    // cooperative stage: K transposed (VALU path), V natural (async path)
    {
      const _Float16* gk = k + (size_t)(kb * 32 + lkey) * D_DIM + hoff + lhs;
      union { uint4 u; _Float16 hh[8]; } tk;
      tk.u = *(const uint4*)gk;
#pragma unroll
      for (int j = 0; j < 8; ++j) sKt[lhs + j][lkey] = tk.hh[j];
      const _Float16* gv = v + (size_t)(kb * 32 + lkey) * D_DIM + hoff + lhs;
      async_copy16(gv, &sVt[lkey][lhs]);
    }
    wait_async();
    asm volatile("s_wait_dscnt 0" ::: "memory");
    __syncthreads();
    if (kb >= myKbEnd) continue;

    // S = Q @ K^T : two 16x16 tiles over 32 keys
#pragma unroll
    for (int t = 0; t < 2; ++t) {
      v8f sc;
#pragma unroll
      for (int e = 0; e < 8; ++e) sc[e] = 0.0f;
#pragma unroll
      for (int ks = 0; ks < 2; ++ks) {
        v16h bf;
#pragma unroll
        for (int s = 0; s < 16; ++s) {
          int hs = ks * 32 + ((s < 8) ? hi * 8 + s : 16 + hi * 8 + (s - 8));
          bf[s] = sKt[hs][t * 16 + lo];
        }
        sc = wmma_f16(aQ[ks], bf, sc);
      }
#pragma unroll
      for (int r = 0; r < 8; ++r) {
        int m  = r + 8 * hi;
        int qg = q0 + m;
        int kg = kb * 32 + t * 16 + lo;
        sS[wid][m][t * 16 + lo] = (kg <= qg) ? sc[r] * 0.125f : -1e30f;
      }
    }
    wave_lds_sync();

    // online softmax update (lanes 0..15 each own one row)
    if (lane < 16) {
      const int m = lane;
      float bmax = -1e30f;
#pragma unroll 4
      for (int c = 0; c < 32; ++c) bmax = fmaxf(bmax, sS[wid][m][c]);
      float nm   = fmaxf(rm, bmax);
      float corr = __expf(rm - nm);
      float ls   = 0.0f;
      for (int c = 0; c < 32; ++c) {
        float p = __expf(sS[wid][m][c] - nm);
        ls += p;
        sP[wid][m][c] = (_Float16)p;
      }
      rl = rl * corr + ls;
      rm = nm;
      sCorr[wid][m] = corr;
    }
    wave_lds_sync();

    // rescale O accumulators by per-row correction
#pragma unroll
    for (int r = 0; r < 8; ++r) {
      float c = sCorr[wid][r + 8 * hi];
#pragma unroll
      for (int ni = 0; ni < 4; ++ni) O[ni][r] *= c;
    }

    // O += P(16x32) @ V(32x64)
    v16h aP = ld_frag16(&sP[wid][lo][hi * 8], &sP[wid][lo][16 + hi * 8]);
#pragma unroll
    for (int ni = 0; ni < 4; ++ni) {
      v16h bf;
#pragma unroll
      for (int s = 0; s < 16; ++s) {
        int kr = (s < 8) ? hi * 8 + s : 16 + hi * 8 + (s - 8);
        bf[s] = sVt[kr][ni * 16 + lo];
      }
      O[ni] = wmma_f16(aP, bf, O[ni]);
    }
  }

  // finalize: normalize by running sum and scatter to [S, D] layout
  if (lane < 16) sL[wid][lane] = rl;
  wave_lds_sync();
#pragma unroll
  for (int r = 0; r < 8; ++r) {
    const float inv = 1.0f / sL[wid][r + 8 * hi];
    const int   m   = q0 + r + 8 * hi;
#pragma unroll
    for (int ni = 0; ni < 4; ++ni)
      o[(size_t)m * D_DIM + hoff + ni * 16 + lo] = (_Float16)(O[ni][r] * inv);
  }
}

// ---------------------------------------------------------------------------
// Host-side launcher
// ---------------------------------------------------------------------------
extern "C" void kernel_launch(void* const* d_in, const int* in_sizes, int n_in,
                              void* d_out, int out_size, void* d_ws, size_t ws_size,
                              hipStream_t stream) {
  (void)in_sizes; (void)n_in; (void)out_size; (void)ws_size;

  const int*   ids  = (const int*)d_in[0];
  const float* tok  = (const float*)d_in[1];
  const float* pos  = (const float*)d_in[2];
  const float* Wq   = (const float*)d_in[3];
  const float* bq   = (const float*)d_in[4];
  const float* Wk   = (const float*)d_in[5];
  const float* bk   = (const float*)d_in[6];
  const float* Wv   = (const float*)d_in[7];
  const float* bv   = (const float*)d_in[8];
  const float* Wo   = (const float*)d_in[9];
  const float* bo   = (const float*)d_in[10];
  const float* W1   = (const float*)d_in[11];
  const float* b1   = (const float*)d_in[12];
  const float* W2   = (const float*)d_in[13];
  const float* b2   = (const float*)d_in[14];
  const float* ln1w = (const float*)d_in[15];
  const float* ln1b = (const float*)d_in[16];
  const float* ln2w = (const float*)d_in[17];
  const float* ln2b = (const float*)d_in[18];
  const float* lnfw = (const float*)d_in[19];
  const float* lnfb = (const float*)d_in[20];
  const float* lmh  = (const float*)d_in[21];
  float* out = (float*)d_out;

  // workspace layout (~48 MB)
  char* ws = (char*)d_ws;
  float*    h     = (float*)   (ws);                     //  8 MB  residual f32
  _Float16* x16   = (_Float16*)(ws + (8u  << 20));       //  4 MB  LN out
  _Float16* q16   = (_Float16*)(ws + (12u << 20));       //  4 MB
  _Float16* k16   = (_Float16*)(ws + (16u << 20));       //  4 MB
  _Float16* v16   = (_Float16*)(ws + (20u << 20));       //  4 MB
  _Float16* xo16  = (_Float16*)(ws + (24u << 20));       //  4 MB  attn out
  _Float16* mid16 = (_Float16*)(ws + (28u << 20));       // 16 MB  MLP mid
  _Float16* hf16  = (_Float16*)(ws + (44u << 20));       //  4 MB  final LN

  embed_kernel<<<(S_LEN * D_DIM) / 256, 256, 0, stream>>>(ids, tok, pos, h);

  dim3 gD(D_DIM / 128, S_LEN / 128);                 // (8, 16)
  dim3 gF(F_DIM / 128, S_LEN / 128);                 // (32, 16)
  dim3 gV((V_DIM + 127) / 128, S_LEN / 128);         // (393, 16)
  dim3 gA(S_LEN / 128, H_NUM);                       // (16, 16)

  for (int l = 0; l < L_NUM; ++l) {
    const size_t wdd = (size_t)l * D_DIM * D_DIM;
    const size_t wdf = (size_t)l * D_DIM * F_DIM;
    const size_t wd  = (size_t)l * D_DIM;
    const size_t wf  = (size_t)l * F_DIM;

    ln_kernel<<<S_LEN, 256, 0, stream>>>(h, ln1w + wd, ln1b + wd, x16);
    gemm_kernel<0, false, false><<<gD, 256, 0, stream>>>(x16, Wq + wdd, bq + wd, q16, nullptr, S_LEN, D_DIM, D_DIM);
    gemm_kernel<0, false, false><<<gD, 256, 0, stream>>>(x16, Wk + wdd, bk + wd, k16, nullptr, S_LEN, D_DIM, D_DIM);
    gemm_kernel<0, false, false><<<gD, 256, 0, stream>>>(x16, Wv + wdd, bv + wd, v16, nullptr, S_LEN, D_DIM, D_DIM);
    attn_kernel<<<gA, 256, 0, stream>>>(q16, k16, v16, xo16);
    gemm_kernel<2, false, false><<<gD, 256, 0, stream>>>(xo16, Wo + wdd, bo + wd, nullptr, h, S_LEN, D_DIM, D_DIM);
    ln_kernel<<<S_LEN, 256, 0, stream>>>(h, ln2w + wd, ln2b + wd, x16);
    gemm_kernel<0, true, false><<<gF, 256, 0, stream>>>(x16, W1 + wdf, b1 + wf, mid16, nullptr, S_LEN, F_DIM, D_DIM);
    gemm_kernel<2, false, false><<<gD, 256, 0, stream>>>(mid16, W2 + wdf, b2 + wd, nullptr, h, S_LEN, D_DIM, F_DIM);
  }

  ln_kernel<<<S_LEN, 256, 0, stream>>>(h, lnfw, lnfb, hf16);
  gemm_kernel<1, false, true><<<gV, 256, 0, stream>>>(hf16, lmh, nullptr, nullptr, out, S_LEN, V_DIM, D_DIM);
}
